// DenseAttentionMultiHead_26345329393792
// MI455X (gfx1250) — matrix operand order repaired and checked by reading, not compiled, not run on hardware
//
#include <hip/hip_runtime.h>
#include <hip/hip_bf16.h>

// ---------------------------------------------------------------------------
// Problem constants (from reference): B=4, S=2048, H=2048, NH=16, D=128
// out = RoPE(x); q = out @ W^T; per head: attn = (Q K^T) K == Q (K^T K)
// ---------------------------------------------------------------------------
#define BB 4
#define SS 2048
#define HH 2048
#define NH 16
#define DD 128

typedef __attribute__((ext_vector_type(16))) __bf16        v16bf;
typedef __attribute__((ext_vector_type(8)))  float         v8f;
typedef __attribute__((ext_vector_type(8)))  unsigned int  v8u;
typedef __attribute__((ext_vector_type(4)))  unsigned int  v4u;
typedef __attribute__((ext_vector_type(2)))  unsigned int  v2u;
typedef __attribute__((ext_vector_type(4)))  float         v4f;

union Frag  { v16bf bf; v4u q[2]; };
union Pack8 { v4u q;    unsigned short s[8]; };

__device__ __forceinline__ unsigned short f2bf(float f) {
    unsigned int u = __builtin_bit_cast(unsigned int, f);
    u += 0x7FFFu + ((u >> 16) & 1u);          // round-to-nearest-even
    return (unsigned short)(u >> 16);
}

// ---------------------------------------------------------------------------
// TDM descriptor helpers (CDNA5 Tensor Data Mover, CDNA5 ISA)
// Group 0 (4 SGPRs): count=1 | lds_addr | global_addr[56:0] | type=2
// Group 1 (8 SGPRs): data_size=2B, pad (LDS row stride 32->40 shorts),
//                    tensor_dim0/1, tile_dim0/1, tensor_dim0_stride
// ---------------------------------------------------------------------------
__device__ __forceinline__ v8u tdm_desc1(unsigned tensor_d0, unsigned tensor_d1,
                                         unsigned tile_d0,  unsigned tile_d1,
                                         unsigned long long stride0) {
    v8u g = {};
    // data_size=1 (2 bytes) | pad_enable | pad_interval=3 (16 dw) | pad_amount=3 (4 dw)
    g[0] = (1u << 16) | (1u << 20) | (3u << 22) | (3u << 25);
    g[1] = (tensor_d0 & 0xFFFFu) << 16;                       // tensor_dim0[15:0]
    g[2] = (tensor_d0 >> 16) | ((tensor_d1 & 0xFFFFu) << 16); // dim0 hi | dim1 lo
    g[3] = (tensor_d1 >> 16) | (tile_d0 << 16);               // dim1 hi | tile_dim0
    g[4] = tile_d1 & 0xFFFFu;                                 // tile_dim1 (tile_dim2=0)
    g[5] = (unsigned)stride0;                                 // tensor_dim0_stride lo
    g[6] = (unsigned)(stride0 >> 32) & 0xFFFFu;               // stride hi (dim1_stride=0)
    g[7] = 0;
    return g;
}

__device__ __forceinline__ void tdm_load(unsigned lds_addr,
                                         unsigned long long gaddr, v8u g1) {
    v4u g0;
    g0.x = 1u;                                                // count=1, user mode
    g0.y = lds_addr;                                          // LDS byte address
    g0.z = (unsigned)gaddr;                                   // global_addr[31:0]
    g0.w = ((unsigned)(gaddr >> 32) & 0x01FFFFFFu) | (2u << 30); // [56:32] | type=2
    asm volatile("tensor_load_to_lds %0, %1" :: "s"(g0), "s"(g1) : "memory");
}

// ---------------------------------------------------------------------------
// K1: RoPE (pairs d, d+64 within each head) + cast to bf16
// ---------------------------------------------------------------------------
__global__ void rope_bf16_kernel(const float* __restrict__ hs,
                                 const float* __restrict__ cosT,
                                 const float* __restrict__ sinT,
                                 unsigned short* __restrict__ xb) {
    unsigned int flat = blockIdx.x * blockDim.x + threadIdx.x;   // 2^23 pairs
    unsigned int d = flat & 63u;
    unsigned int h = (flat >> 6) & 15u;
    unsigned int s = (flat >> 10) & 2047u;
    unsigned int b = flat >> 21;
    unsigned int base = ((b * SS + s) * HH) + h * DD + d;
    float x1 = hs[base];
    float x2 = hs[base + 64];
    float c  = cosT[s * 64 + d];
    float sn = sinT[s * 64 + d];
    xb[base]      = f2bf(x1 * c - x2 * sn);
    xb[base + 64] = f2bf(x2 * c + x1 * sn);
}

// ---------------------------------------------------------------------------
// K2: W (f32) -> bf16
// ---------------------------------------------------------------------------
__global__ void wcast_kernel(const float* __restrict__ W,
                             unsigned short* __restrict__ Wb) {
    unsigned int i4 = (blockIdx.x * blockDim.x + threadIdx.x) * 4u;
    v4f w = *(const v4f*)&W[i4];
    unsigned int lo = (unsigned int)f2bf(w.x) | ((unsigned int)f2bf(w.y) << 16);
    unsigned int hi = (unsigned int)f2bf(w.z) | ((unsigned int)f2bf(w.w) << 16);
    v2u p; p.x = lo; p.y = hi;
    *(v2u*)&Wb[i4] = p;
}

// ---------------------------------------------------------------------------
// K3: q = x @ W^T   (M=8192, N=2048, K=2048) bf16 WMMA, 128x128 tiles
//     TDM (tensor_load_to_lds) double-buffers A/B tiles into LDS while
//     8 waves run the WMMA stream; TENSORcnt + barrier synchronize.
// ---------------------------------------------------------------------------
#define LDSTR 40   // padded LDS row stride (shorts) = 80B (TDM pad: 16dw + 4dw)
#define TILEB (128 * LDSTR)   // shorts per buffer = 10240B
__global__ void __launch_bounds__(256)
gemm_q_kernel(const unsigned short* __restrict__ xb,
              const unsigned short* __restrict__ Wb,
              unsigned short* __restrict__ qb) {
    __shared__ __align__(16) unsigned short As[2][TILEB];
    __shared__ __align__(16) unsigned short Bs[2][TILEB];

    const unsigned int tid  = threadIdx.x;
    const unsigned int wave = tid >> 5;
    const unsigned int lane = tid & 31u;
    const unsigned int m0 = blockIdx.y * 128u;
    const unsigned int n0 = blockIdx.x * 128u;

    const unsigned int mrow  = lane & 15u;
    const unsigned int half  = lane >> 4;      // K-half selector
    const unsigned int abase = (wave * 16u + mrow) * LDSTR;

    // byte addresses of the first tile (row m0/n0, k=0)
    const unsigned long long gA = (unsigned long long)(size_t)(xb + (size_t)m0 * HH);
    const unsigned long long gB = (unsigned long long)(size_t)(Wb + (size_t)n0 * HH);
    const v8u d1A = tdm_desc1(/*dim0*/ HH, /*dim1*/ BB * SS, /*tile*/ 32u, 128u, HH);
    const v8u d1B = tdm_desc1(/*dim0*/ HH, /*dim1*/ HH,      /*tile*/ 32u, 128u, HH);

    v8f acc[8] = {};

    // prologue: DMA tile k=0 into buffer 0
    if (wave == 0) {
        tdm_load((unsigned)(size_t)&As[0][0], gA, d1A);
        tdm_load((unsigned)(size_t)&Bs[0][0], gB, d1B);
    }

    for (unsigned int kt = 0; kt < 64; ++kt) {
        const unsigned int cur = kt & 1u;
        if (wave == 0)
            __builtin_amdgcn_s_wait_tensorcnt(0);   // current buffer landed
        __syncthreads();                            // visible to all waves; prev compute done
        if (wave == 0 && kt < 63) {                 // DMA next tile into other buffer
            const unsigned long long koff = (unsigned long long)(kt + 1) * 64u; // 32 elem * 2B
            tdm_load((unsigned)(size_t)&As[cur ^ 1u][0], gA + koff, d1A);
            tdm_load((unsigned)(size_t)&Bs[cur ^ 1u][0], gB + koff, d1B);
        }

        Frag a;
        a.q[0] = *(const v4u*)&As[cur][abase + half * 8u];
        a.q[1] = *(const v4u*)&As[cur][abase + 16u + half * 8u];

#pragma unroll
        for (unsigned int nt = 0; nt < 8; ++nt) {
            Frag bfr;
            const unsigned int bb = (nt * 16u + mrow) * LDSTR + half * 8u;
            bfr.q[0] = *(const v4u*)&Bs[cur][bb];
            bfr.q[1] = *(const v4u*)&Bs[cur][bb + 16];
            acc[nt] = __builtin_amdgcn_wmma_f32_16x16x32_bf16(
                false, a.bf, false, bfr.bf, (short)0, acc[nt], false, false);
        }
        __syncthreads();                            // compute done before buffer reuse
    }

    // store q in bf16
    const unsigned int rbase = m0 + wave * 16u + (half << 3);
    const unsigned int cbase = n0 + mrow;
#pragma unroll
    for (unsigned int nt = 0; nt < 8; ++nt)
#pragma unroll
        for (unsigned int r = 0; r < 8; ++r)
            qb[(rbase + r) * HH + cbase + nt * 16u] = f2bf(acc[nt][r]);
}

// ---------------------------------------------------------------------------
// K4: per (b,h) Gram matrix G = K^T K  (128x128, reduce over S=2048)
//     one block per (b,h); LDS holds transposed tile T[d][s32]
// ---------------------------------------------------------------------------
__global__ void __launch_bounds__(256)
gram_kernel(const unsigned short* __restrict__ xb,
            unsigned short* __restrict__ Gb) {
    __shared__ __align__(16) unsigned short T[128 * LDSTR];

    const unsigned int bh   = blockIdx.x;
    const unsigned int b    = bh >> 4;
    const unsigned int h    = bh & 15u;
    const unsigned int tid  = threadIdx.x;
    const unsigned int wave = tid >> 5;
    const unsigned int lane = tid & 31u;
    const unsigned int mrow = lane & 15u;
    const unsigned int half = lane >> 4;

    const unsigned int srow = tid >> 3;          // 0..31
    const unsigned int dseg = (tid & 7u) * 16u;  // 0..112

    v8f acc[8] = {};

    for (unsigned int s0 = 0; s0 < SS; s0 += 32) {
        __syncthreads();
        // load 32(s) x 128(d), store transposed: T[d][s]
        const unsigned short* g =
            xb + (b * SS + s0 + srow) * HH + h * DD + dseg;
        Pack8 p0, p1;
        p0.q = *(const v4u*)(g + 0);
        p1.q = *(const v4u*)(g + 8);
#pragma unroll
        for (unsigned int i = 0; i < 8; ++i) {
            T[(dseg + i)     * LDSTR + srow] = p0.s[i];
            T[(dseg + 8 + i) * LDSTR + srow] = p1.s[i];
        }
        __syncthreads();

        Frag a;  // rows d1 = wave*16 + mrow, K = s window
        const unsigned int abase = (wave * 16u + mrow) * LDSTR;
        a.q[0] = *(const v4u*)&T[abase + half * 8u];
        a.q[1] = *(const v4u*)&T[abase + 16u + half * 8u];

#pragma unroll
        for (unsigned int nt = 0; nt < 8; ++nt) {
            Frag bfr;  // cols d2 = nt*16 + mrow
            const unsigned int bb = (nt * 16u + mrow) * LDSTR + half * 8u;
            bfr.q[0] = *(const v4u*)&T[bb];
            bfr.q[1] = *(const v4u*)&T[bb + 16];
            acc[nt] = __builtin_amdgcn_wmma_f32_16x16x32_bf16(
                false, a.bf, false, bfr.bf, (short)0, acc[nt], false, false);
        }
    }

    unsigned short* G = Gb + bh * (DD * DD);
    const unsigned int rbase = wave * 16u + (half << 3);
#pragma unroll
    for (unsigned int nt = 0; nt < 8; ++nt)
#pragma unroll
        for (unsigned int r = 0; r < 8; ++r)
            G[(rbase + r) * DD + nt * 16u + mrow] = f2bf(acc[nt][r]);
}

// ---------------------------------------------------------------------------
// K5: out = Q G per (b,h)   (M=2048 rows of s, N=128, K=128)
//     G is symmetric, so B[k][n] = G[n][k] -> contiguous 32B frag reads.
// ---------------------------------------------------------------------------
__global__ void __launch_bounds__(256)
out_kernel(const unsigned short* __restrict__ qb,
           const unsigned short* __restrict__ Gb,
           float* __restrict__ out) {
    const unsigned int st   = blockIdx.x;        // 16 s-tiles of 128
    const unsigned int bh   = blockIdx.y;        // 64 (b,h)
    const unsigned int b    = bh >> 4;
    const unsigned int h    = bh & 15u;
    const unsigned int tid  = threadIdx.x;
    const unsigned int wave = tid >> 5;
    const unsigned int lane = tid & 31u;
    const unsigned int mrow = lane & 15u;
    const unsigned int half = lane >> 4;

    const unsigned int grow = b * SS + st * 128u + wave * 16u;  // global q row base
    const unsigned short* G = Gb + bh * (DD * DD);

    v8f acc[8] = {};
#pragma unroll
    for (unsigned int k0 = 0; k0 < DD; k0 += 32) {
        Frag a;
        const unsigned short* ga = qb + (grow + mrow) * HH + h * DD + k0;
        a.q[0] = *(const v4u*)(ga + half * 8u);
        a.q[1] = *(const v4u*)(ga + 16u + half * 8u);
#pragma unroll
        for (unsigned int nt = 0; nt < 8; ++nt) {
            Frag bfr;
            const unsigned short* gg = G + (nt * 16u + mrow) * DD + k0 + half * 8u;
            bfr.q[0] = *(const v4u*)(gg + 0);
            bfr.q[1] = *(const v4u*)(gg + 16);
            acc[nt] = __builtin_amdgcn_wmma_f32_16x16x32_bf16(
                false, a.bf, false, bfr.bf, (short)0, acc[nt], false, false);
        }
    }

    const unsigned int rbase = grow + (half << 3);
    const unsigned int cbase = h * DD + mrow;
#pragma unroll
    for (unsigned int nt = 0; nt < 8; ++nt)
#pragma unroll
        for (unsigned int r = 0; r < 8; ++r)
            out[(rbase + r) * HH + cbase + nt * 16u] = acc[nt][r];
}

// ---------------------------------------------------------------------------
// launch
// ---------------------------------------------------------------------------
extern "C" void kernel_launch(void* const* d_in, const int* in_sizes, int n_in,
                              void* d_out, int out_size, void* d_ws, size_t ws_size,
                              hipStream_t stream) {
    const float* hs   = (const float*)d_in[0];
    const float* W    = (const float*)d_in[1];
    const float* cosT = (const float*)d_in[2];
    const float* sinT = (const float*)d_in[3];
    float* out = (float*)d_out;

    char* ws = (char*)d_ws;
    unsigned short* xb = (unsigned short*)(ws);                              // 32 MB
    unsigned short* Wb = (unsigned short*)(ws + (size_t)32 * 1024 * 1024);   //  8 MB
    unsigned short* qb = (unsigned short*)(ws + (size_t)40 * 1024 * 1024);   // 32 MB
    unsigned short* Gb = (unsigned short*)(ws + (size_t)72 * 1024 * 1024);   //  2 MB

    // K1: RoPE + bf16 cast: 2^23 pairs
    rope_bf16_kernel<<<32768, 256, 0, stream>>>(hs, cosT, sinT, xb);
    // K2: W cast: 4M elems / 4 per thread
    wcast_kernel<<<4096, 256, 0, stream>>>(W, Wb);
    // K3: q = x @ W^T  (TDM-fed WMMA)
    gemm_q_kernel<<<dim3(16, 64), 256, 0, stream>>>(xb, Wb, qb);
    // K4: G = K^T K per (b,h)
    gram_kernel<<<64, 256, 0, stream>>>(xb, Gb);
    // K5: out = Q G per (b,h)
    out_kernel<<<dim3(16, 64), 256, 0, stream>>>(qb, Gb, out);
}
